// newResNet_76115410419910
// MI455X (gfx1250) — matrix operand order repaired and checked by reading, not compiled, not run on hardware
//
#include <hip/hip_runtime.h>
#include <hip/hip_bf16.h>

// ResNet-20 (CIFAR, option-A shortcuts) forward, batch 256. The broadcast 3D
// stem is folded algebraically into a 3x3 2D conv. Every conv and the final FC
// run as GEMMs on the CDNA5 WMMA pipe (v_wmma_f32_16x16x32_bf16, bf16 A/B,
// f32 accumulate). Pipeline per conv:
//   im2col_kernel : gather NCHW bf16 acts -> A panel [rows][Kpad] (chunked)
//   gemm kernel   : weights in LDS; one wave owns a 16 x Cout strip, A
//                   fragment loaded once per K-block and reused across all
//                   N-tiles (KBLOCKS x NTILES WMMAs, fully unrolled);
//                   BN sum/sumsq fused via f32 atomics
//   bn_apply      : batch-norm + option-A shortcut + ReLU -> bf16 acts
//
// Workspace layout (bytes from d_ws, ~80 MiB):
//   [0,        8Mi)   ACT_A bf16 activations (ping)
//   [8Mi,     16Mi)   ACT_B bf16 activations (block-mid / x staging)
//   [16Mi,    24Mi)   ACT_C bf16 activations (pong)
//   [24Mi,    40Mi)   CF32  f32 conv output
//   [40Mi,  +512)     STATS per-channel sum/sumsq
//   [40Mi+4K, +128K)  WBUF  bf16 packed weights [Cout][Kpad]
//   [40Mi+256K,+32K)  POOL  bf16 pooled features [256][64]
//   [41Mi,    79Mi)   ASTAGE bf16 im2col panel (chunked, <= 37.8 MiB)

typedef __attribute__((ext_vector_type(16))) __bf16 v16bf;
typedef __attribute__((ext_vector_type(8)))  __bf16 v8bf;
typedef __attribute__((ext_vector_type(8)))  float  v8f;

#define LOWD 40
#define BATCH 256
#define CHUNK_ROWS 65536          // M-chunk for the im2col staging buffer

__device__ __forceinline__ unsigned short f32_to_bf16_bits(float f) {
  unsigned int u = __builtin_bit_cast(unsigned int, f);
  u += 0x7FFFu + ((u >> 16) & 1u);             // round-to-nearest-even
  return (unsigned short)(u >> 16);
}
__device__ __forceinline__ float bf16_bits_to_f32(unsigned short h) {
  return __builtin_bit_cast(float, (unsigned int)h << 16);
}

// ---------------------------------------------------------------------------
// Prep kernels
// ---------------------------------------------------------------------------
__global__ void convert_f32_to_bf16_kernel(const float* __restrict__ in,
                                           unsigned short* __restrict__ out,
                                           int n) {
  int i = blockIdx.x * blockDim.x + threadIdx.x;
  if (i < n) out[i] = f32_to_bf16_bits(in[i]);
}

// Fold the broadcast 3D stem into an effective 3x3 conv weight, bf16 [16][32].
__global__ void stem_weff_kernel(const float* __restrict__ w3d,
                                 const float* __restrict__ a,
                                 unsigned short* __restrict__ wout) {
  int t = threadIdx.x;                     // 512 threads, 1 block
  if (t >= 16 * 32) return;
  int o = t >> 5, k = t & 31;
  float val = 0.0f;
  if (k < 27) {
    int ci = k / 9;
    int r  = k - ci * 9;
    int kh = r / 3, kw = r - kh * 3;
    for (int kd = 0; kd < LOWD; ++kd) {
      int lo = 19 - kd; if (lo < 0) lo = 0;
      int hi = 58 - kd; if (hi > 39) hi = 39;
      float A = 0.0f;
      for (int d = lo; d <= hi; ++d) A += a[d];
      val += w3d[(((o * 3 + ci) * LOWD + kd) * 3 + kh) * 3 + kw] * A;
    }
    val *= (1.0f / (float)LOWD);
  }
  wout[(o << 5) | k] = f32_to_bf16_bits(val);
}

__global__ void convert_w_kernel(const float* __restrict__ w,
                                 unsigned short* __restrict__ wout,
                                 int Cout, int Kreal, int Kpad) {
  int i = blockIdx.x * blockDim.x + threadIdx.x;
  if (i >= Cout * Kpad) return;
  int o = i / Kpad, k = i - o * Kpad;
  wout[i] = (k < Kreal) ? f32_to_bf16_bits(w[o * Kreal + k]) : (unsigned short)0;
}

// FC weights [10][64] -> bf16 [16][64], rows 10..15 zeroed.
__global__ void convert_fc_kernel(const float* __restrict__ w,
                                  unsigned short* __restrict__ wout) {
  int i = blockIdx.x * blockDim.x + threadIdx.x;
  if (i >= 16 * 64) return;
  int o = i >> 6, k = i & 63;
  wout[i] = (o < 10) ? f32_to_bf16_bits(w[o * 64 + k]) : (unsigned short)0;
}

__global__ void zero_stats_kernel(float* __restrict__ stats) {
  if (threadIdx.x < 128) stats[threadIdx.x] = 0.0f;
}

// ---------------------------------------------------------------------------
// im2col: bf16 NCHW activations -> A panel [rows][Kpad] for one M-chunk.
// k = ci*9 + kh*3 + kw ; m = b*HWout + oh*Wout + ow (global row = m0 + local)
// ---------------------------------------------------------------------------
__global__ void im2col_kernel(const unsigned short* __restrict__ act,
                              unsigned short* __restrict__ Astage,
                              int m0, int rows, int Cin, int Hin, int Win,
                              int HWout, int Wout, int stride,
                              int Kreal, int Kpad) {
  int i = blockIdx.x * blockDim.x + threadIdx.x;
  if (i >= rows * Kpad) return;
  const int ml = i / Kpad;
  const int k  = i - ml * Kpad;
  unsigned short bits = 0;
  if (k < Kreal) {
    const int m   = m0 + ml;
    const int b   = m / HWout;
    const int rhw = m - b * HWout;
    const int oh  = rhw / Wout;
    const int ow  = rhw - oh * Wout;
    const int ci  = k / 9;
    const int r9  = k - ci * 9;
    const int kh  = r9 / 3;
    const int kw  = r9 - kh * 3;
    const int ih  = oh * stride + kh - 1;
    const int iw  = ow * stride + kw - 1;
    if ((unsigned)ih < (unsigned)Hin && (unsigned)iw < (unsigned)Win)
      bits = act[((b * Cin + ci) * Hin + ih) * Win + iw];
  }
  Astage[i] = bits;
}

// ---------------------------------------------------------------------------
// GEMM on WMMA: D[rows x Cout] = A[rows x Kpad] * W^T.
// One wave owns a full 16 x Cout strip (NTILES = Cout/16 accumulators): the A
// fragment is loaded once per K-block (2x b128) and reused for all N-tiles;
// B fragments stream from LDS (32B ds reads). K and N loops fully unrolled.
// BN sum/sumsq fused via f32 atomics.
// ---------------------------------------------------------------------------
template <int KBLOCKS, int NTILES>
__global__ void gemm_conv_wmma_kernel(const unsigned short* __restrict__ Astage,
                                      const unsigned short* __restrict__ wts,
                                      float* __restrict__ out,
                                      float* __restrict__ stats,
                                      int m0, int rows,
                                      int HW, int Hout, int Wout) {
  constexpr int Kpad = KBLOCKS * 32;
  constexpr int Cout = NTILES * 16;
  extern __shared__ unsigned short wlds[];       // Cout * Kpad bf16

  // Cooperative weight stage: global -> LDS (16B vector copies).
  constexpr int totalW = Cout * Kpad;
  for (int i = threadIdx.x * 8; i < totalW; i += blockDim.x * 8)
    *reinterpret_cast<uint4*>(&wlds[i]) =
        *reinterpret_cast<const uint4*>(wts + i);
  __syncthreads();

  const int lane = threadIdx.x & 31;
  const int half = lane >> 4;
  const int l15  = lane & 15;
  const int mtiles = rows >> 4;

  int wave   = (int)((blockIdx.x * blockDim.x + threadIdx.x) >> 5);
  int nwaves = (int)((gridDim.x * blockDim.x) >> 5);

  for (int mt = wave; mt < mtiles; mt += nwaves) {
    const unsigned short* arow = Astage + ((mt << 4) | l15) * Kpad;
    const unsigned short* wcol = &wlds[l15 * Kpad + (half << 4)];

    v8f acc[NTILES];
#pragma unroll
    for (int nt = 0; nt < NTILES; ++nt) acc[nt] = 0.0f;

#pragma unroll
    for (int kb = 0; kb < KBLOCKS; ++kb) {
      // A fragment: lane row l15, K chunk [kb*32+half*16, +16).
      const v8bf alo = *reinterpret_cast<const v8bf*>(arow + (kb << 5) + (half << 4));
      const v8bf ahi = *reinterpret_cast<const v8bf*>(arow + (kb << 5) + (half << 4) + 8);
      const v16bf afrag = __builtin_shufflevector(
          alo, ahi, 0, 1, 2, 3, 4, 5, 6, 7, 8, 9, 10, 11, 12, 13, 14, 15);
      if (kb + 1 < KBLOCKS)
        __builtin_prefetch(arow + ((kb + 1) << 5) + (half << 3), 0, 1);
      // Reuse A across all N-tiles; B streams from LDS (32B per fragment).
#pragma unroll
      for (int nt = 0; nt < NTILES; ++nt) {
        const v16bf bfrag = *reinterpret_cast<const v16bf*>(
            wcol + nt * 16 * Kpad + (kb << 5));
        acc[nt] = __builtin_amdgcn_wmma_f32_16x16x32_bf16(
            false, afrag, false, bfrag, (short)0, acc[nt], false, false);
      }
    }

    // Epilogue: lane owns channel n = nt*16 + l15, rows half*8 + r.
#pragma unroll
    for (int nt = 0; nt < NTILES; ++nt) {
      const int n = (nt << 4) | l15;
      float s = 0.0f, s2 = 0.0f;
#pragma unroll
      for (int r = 0; r < 8; ++r) {
        const float v  = acc[nt][r];
        const int  mg  = m0 + (mt << 4) + (half << 3) + r;   // global M row
        const int  bb  = mg / HW;
        const int  rr  = mg - bb * HW;
        const int  oh2 = rr / Wout;
        const int  ow2 = rr - oh2 * Wout;
        out[((bb * Cout + n) * Hout + oh2) * Wout + ow2] = v;
        s += v;
        s2 += v * v;
      }
      atomicAdd(&stats[n], s);
      atomicAdd(&stats[Cout + n], s2);
    }
  }
}

// ---------------------------------------------------------------------------
// BN (batch stats) + optional option-A shortcut + ReLU -> bf16 activation.
//   scmode 0: none   1: identity add   2: stride-2 subsample + channel pad
// ---------------------------------------------------------------------------
__global__ void bn_apply_kernel(const float* __restrict__ conv,
                                const float* __restrict__ stats,
                                const float* __restrict__ gamma,
                                const float* __restrict__ beta,
                                unsigned short* __restrict__ actout,
                                const unsigned short* __restrict__ scin,
                                int scmode, int C, int H, int W,
                                int inC, int padc, float invcount) {
  int i = blockIdx.x * blockDim.x + threadIdx.x;
  const int total = BATCH * C * H * W;
  if (i >= total) return;
  int w = i % W;
  int t = i / W;
  int h = t % H; t /= H;
  int c = t % C;
  int b = t / C;

  const float mean = stats[c] * invcount;
  const float var  = stats[C + c] * invcount - mean * mean;
  float y = (conv[i] - mean) * rsqrtf(var + 1e-5f) * gamma[c] + beta[c];

  if (scmode == 1) {
    y += bf16_bits_to_f32(scin[i]);
  } else if (scmode == 2) {
    const int cin = c - padc;
    if (cin >= 0 && cin < inC) {
      const int Hin2 = H * 2, Win2 = W * 2;
      y += bf16_bits_to_f32(scin[((b * inC + cin) * Hin2 + h * 2) * Win2 + w * 2]);
    }
  }
  actout[i] = f32_to_bf16_bits(fmaxf(y, 0.0f));
}

// ---------------------------------------------------------------------------
// Global average pool: act bf16 [256][64][8][8] -> pooled bf16 [256][64]
// ---------------------------------------------------------------------------
__global__ void pool_kernel(const unsigned short* __restrict__ act,
                            unsigned short* __restrict__ pooled) {
  const int b = blockIdx.x;
  const int c = threadIdx.x;            // 64 threads
  const unsigned short* p = act + (b * 64 + c) * 64;
  float acc = 0.0f;
#pragma unroll
  for (int i = 0; i < 64; ++i) acc += bf16_bits_to_f32(p[i]);
  pooled[b * 64 + c] = f32_to_bf16_bits(acc * (1.0f / 64.0f));
}

// ---------------------------------------------------------------------------
// FC 64->10 on WMMA: [256 x 64] x [64 x 16] (cols 10..15 junk, not stored).
// 16 M-tiles, 2 K-blocks; bias fused in epilogue.
// ---------------------------------------------------------------------------
__global__ void fc_wmma_kernel(const unsigned short* __restrict__ pooled,
                               const unsigned short* __restrict__ wfcb,
                               const float* __restrict__ bfc,
                               float* __restrict__ out) {
  const int lane = threadIdx.x & 31;
  const int half = lane >> 4;
  const int l15  = lane & 15;
  const int mt   = (int)((blockIdx.x * blockDim.x + threadIdx.x) >> 5);
  if (mt >= 16) return;                 // wave-uniform

  const unsigned short* arow = pooled + ((mt << 4) | l15) * 64;
  v8f c = 0.0f;
#pragma unroll
  for (int kb = 0; kb < 2; ++kb) {
    const v8bf alo = *reinterpret_cast<const v8bf*>(arow + (kb << 5) + (half << 4));
    const v8bf ahi = *reinterpret_cast<const v8bf*>(arow + (kb << 5) + (half << 4) + 8);
    const v16bf afrag = __builtin_shufflevector(
        alo, ahi, 0, 1, 2, 3, 4, 5, 6, 7, 8, 9, 10, 11, 12, 13, 14, 15);
    const v16bf bfrag = *reinterpret_cast<const v16bf*>(
        wfcb + l15 * 64 + (kb << 5) + (half << 4));
    c = __builtin_amdgcn_wmma_f32_16x16x32_bf16(
        false, afrag, false, bfrag, (short)0, c, false, false);
  }
  const int n = l15;
  if (n < 10) {
    const float bias = bfc[n];
#pragma unroll
    for (int r = 0; r < 8; ++r) {
      const int m = (mt << 4) + (half << 3) + r;
      out[m * 10 + n] = c[r] + bias;
    }
  }
}

// ---------------------------------------------------------------------------
// Host orchestration
// ---------------------------------------------------------------------------
static inline int roundup32(int k) { return (k + 31) & ~31; }

static void launch_gemm(int kblocks, int ntiles, const unsigned short* A,
                        const unsigned short* wts, float* out, float* stats,
                        int m0, int rows, int HW, int Hout, int Wout,
                        hipStream_t stream) {
  const int tiles = rows / 16;                   // one 16 x Cout strip per wave
  int blocks = (tiles + 7) / 8;                  // 256 thr = 8 waves/block
  if (blocks > 8192) blocks = 8192;
  const size_t ldsB = (size_t)(ntiles * 16) * (size_t)(kblocks * 32) * 2;
#define GEMM_CASE(KB, NT)                                                    \
  if (kblocks == KB && ntiles == NT) {                                       \
    gemm_conv_wmma_kernel<KB, NT><<<blocks, 256, ldsB, stream>>>(            \
        A, wts, out, stats, m0, rows, HW, Hout, Wout);                       \
    return;                                                                  \
  }
  GEMM_CASE(1, 1)     // stem:      K=32,  Cout=16
  GEMM_CASE(5, 1)     // stage1:    K=160, Cout=16
  GEMM_CASE(5, 2)     // s2 conv1:  K=160, Cout=32
  GEMM_CASE(9, 2)     // s2 conv2:  K=288, Cout=32
  GEMM_CASE(9, 4)     // s3 conv1:  K=288, Cout=64
  GEMM_CASE(18, 4)    // s3 conv2:  K=576, Cout=64
#undef GEMM_CASE
}

// One conv layer: weights->bf16, stats=0, chunked (im2col ; GEMM), on stream.
static void run_conv(const unsigned short* actin, const float* w,
                     const unsigned short* wbuf_pre /*non-null: packed wts*/,
                     unsigned short* WBUF, unsigned short* ASTAGE,
                     float* CF, float* STATS,
                     int Cin, int Hin, int Win, int Cout, int Hout, int Wout,
                     int stride, hipStream_t stream) {
  const int Kreal = (Cin == 3) ? 27 : Cin * 9;
  const int Kpad  = roundup32(Kreal);
  const unsigned short* wts = wbuf_pre;
  if (!wbuf_pre) {
    convert_w_kernel<<<(Cout * Kpad + 255) / 256, 256, 0, stream>>>(
        w, WBUF, Cout, Kreal, Kpad);
    wts = WBUF;
  }
  zero_stats_kernel<<<1, 128, 0, stream>>>(STATS);
  const int HW = Hout * Wout;
  const int M  = BATCH * HW;
  for (int m0 = 0; m0 < M; m0 += CHUNK_ROWS) {
    const int rows = (M - m0 < CHUNK_ROWS) ? (M - m0) : CHUNK_ROWS;
    const int nel  = rows * Kpad;
    im2col_kernel<<<(nel + 255) / 256, 256, 0, stream>>>(
        actin, ASTAGE, m0, rows, Cin, Hin, Win, HW, Wout, stride, Kreal, Kpad);
    launch_gemm(Kpad / 32, Cout / 16, ASTAGE, wts, CF, STATS, m0, rows, HW,
                Hout, Wout, stream);
  }
}

extern "C" void kernel_launch(void* const* d_in, const int* in_sizes, int n_in,
                              void* d_out, int out_size, void* d_ws, size_t ws_size,
                              hipStream_t stream) {
  (void)in_sizes; (void)n_in; (void)out_size; (void)ws_size;
  // Input order: x, w3d, a, g_bn1, b_bn1, 9x(w1,g1,b1,w2,g2,b2), w_fc, b_fc.
  const float* x     = (const float*)d_in[0];
  const float* w3d   = (const float*)d_in[1];
  const float* a     = (const float*)d_in[2];
  const float* g_bn1 = (const float*)d_in[3];
  const float* b_bn1 = (const float*)d_in[4];
  const float* wfc   = (const float*)d_in[59];
  const float* bfc   = (const float*)d_in[60];
  float* out = (float*)d_out;

  char* ws = (char*)d_ws;
  unsigned short* ACTA   = (unsigned short*)(ws);
  unsigned short* ACTB   = (unsigned short*)(ws + (size_t)(8u  << 20));
  unsigned short* ACTC   = (unsigned short*)(ws + (size_t)(16u << 20));
  float*          CF     = (float*)         (ws + (size_t)(24u << 20));
  float*          STATS  = (float*)         (ws + (size_t)(40u << 20));
  unsigned short* WBUF   = (unsigned short*)(ws + (size_t)(40u << 20) + 4096);
  unsigned short* POOL   = (unsigned short*)(ws + (size_t)(40u << 20) + (256u << 10));
  unsigned short* ASTAGE = (unsigned short*)(ws + (size_t)(41u << 20));

  // ---- Stem: folded 3x3 conv (3->16) + BN + ReLU ----
  {
    const int nx = BATCH * 3 * 32 * 32;
    convert_f32_to_bf16_kernel<<<(nx + 255) / 256, 256, 0, stream>>>(x, ACTB, nx);
    stem_weff_kernel<<<1, 512, 0, stream>>>(w3d, a, WBUF);
    run_conv(ACTB, nullptr, WBUF, WBUF, ASTAGE, CF, STATS,
             3, 32, 32, 16, 32, 32, 1, stream);
    const int nt = BATCH * 16 * 32 * 32;
    bn_apply_kernel<<<(nt + 255) / 256, 256, 0, stream>>>(
        CF, STATS, g_bn1, b_bn1, ACTA, nullptr, 0, 16, 32, 32, 0, 0,
        1.0f / (float)(BATCH * 32 * 32));
  }

  // ---- 9 basic blocks ----
  static const int planes_cfg[3]  = {16, 32, 64};
  static const int stride0_cfg[3] = {1, 2, 2};
  unsigned short* cin  = ACTA;
  unsigned short* cout = ACTC;
  int inC = 16, H = 32, W = 32;

  for (int s = 0; s < 3; ++s) {
    for (int blk = 0; blk < 3; ++blk) {
      const int planes = planes_cfg[s];
      const int stride = (blk == 0) ? stride0_cfg[s] : 1;
      const int Ho = H / stride, Wo = W / stride;
      const int base = 5 + (s * 3 + blk) * 6;
      const float* w1 = (const float*)d_in[base + 0];
      const float* g1 = (const float*)d_in[base + 1];
      const float* b1 = (const float*)d_in[base + 2];
      const float* w2 = (const float*)d_in[base + 3];
      const float* g2 = (const float*)d_in[base + 4];
      const float* b2 = (const float*)d_in[base + 5];
      const float invcnt = 1.0f / (float)(BATCH * Ho * Wo);
      const int ntot = BATCH * planes * Ho * Wo;

      // conv1 (stride) + BN + ReLU -> ACTB
      run_conv(cin, w1, nullptr, WBUF, ASTAGE, CF, STATS,
               inC, H, W, planes, Ho, Wo, stride, stream);
      bn_apply_kernel<<<(ntot + 255) / 256, 256, 0, stream>>>(
          CF, STATS, g1, b1, ACTB, nullptr, 0, planes, Ho, Wo, 0, 0, invcnt);

      // conv2 (stride 1) + BN + shortcut + ReLU -> cout
      run_conv(ACTB, w2, nullptr, WBUF, ASTAGE, CF, STATS,
               planes, Ho, Wo, planes, Ho, Wo, 1, stream);
      const int scmode = (stride != 1 || inC != planes) ? 2 : 1;
      bn_apply_kernel<<<(ntot + 255) / 256, 256, 0, stream>>>(
          CF, STATS, g2, b2, cout, cin, scmode, planes, Ho, Wo,
          inC, planes / 4, invcnt);

      unsigned short* tmp = cin; cin = cout; cout = tmp;
      inC = planes; H = Ho; W = Wo;
    }
  }

  // ---- avgpool + FC(64->10) on WMMA ----
  pool_kernel<<<BATCH, 64, 0, stream>>>(cin, POOL);
  convert_fc_kernel<<<4, 256, 0, stream>>>(wfc, WBUF);
  fc_wmma_kernel<<<2, 256, 0, stream>>>(POOL, WBUF, bfc, out);
}
